// TemporalGNNWikiMaths_51247549775953
// MI455X (gfx1250) — matrix operand order repaired and checked by reading, not compiled, not run on hardware
//
#include <hip/hip_runtime.h>

typedef __attribute__((ext_vector_type(16))) _Float16 v16h;
typedef __attribute__((ext_vector_type(8)))  float    v8f;

#define DIM 128
#define GDIM 384   // 3*DIM

// ---------- helpers ----------

__device__ __forceinline__ unsigned key_of(float f) {
  unsigned u = __float_as_uint(f);
  return (u & 0x80000000u) ? ~u : (u | 0x80000000u);
}
__device__ __forceinline__ float key_to_float(unsigned k) {
  unsigned b = (k & 0x80000000u) ? (k & 0x7FFFFFFFu) : ~k;
  return __uint_as_float(b);
}

struct frag2 { v16h hi, lo; };

__device__ __forceinline__ void split1(float a, v16h& h, v16h& l, int e) {
  _Float16 hh = (_Float16)a;
  h[e] = hh;
  l[e] = (_Float16)(a - (float)hh);
}

// A-fragment (16x32 f16, split hi/lo) from f32 row-major source.
// rowp = &A[row*lda + kbase]; lane half = (lane>>4).
// e : K = half*16 + e (16 consecutive floats at rowp + half*16)
__device__ __forceinline__ frag2 load_a_frag_split(const float* rowp, int half) {
  const float* ap = rowp + half * 16;
  float4 a0 = *(const float4*)(ap + 0);
  float4 a1 = *(const float4*)(ap + 4);
  float4 a2 = *(const float4*)(ap + 8);
  float4 a3 = *(const float4*)(ap + 12);
  frag2 r;
  split1(a0.x, r.hi, r.lo, 0);  split1(a0.y, r.hi, r.lo, 1);
  split1(a0.z, r.hi, r.lo, 2);  split1(a0.w, r.hi, r.lo, 3);
  split1(a1.x, r.hi, r.lo, 4);  split1(a1.y, r.hi, r.lo, 5);
  split1(a1.z, r.hi, r.lo, 6);  split1(a1.w, r.hi, r.lo, 7);
  split1(a2.x, r.hi, r.lo, 8);  split1(a2.y, r.hi, r.lo, 9);
  split1(a2.z, r.hi, r.lo, 10); split1(a2.w, r.hi, r.lo, 11);
  split1(a3.x, r.hi, r.lo, 12); split1(a3.y, r.hi, r.lo, 13);
  split1(a3.z, r.hi, r.lo, 14); split1(a3.w, r.hi, r.lo, 15);
  return r;
}

// B-fragment (32x16 f16, split hi/lo) from f32 source; column stored contiguously in K.
// colp = &Bt[col*ldb + kbase + half*16] (16 consecutive K values for this lane)
__device__ __forceinline__ frag2 load_b_frag_split(const float* colp) {
  float4 b0 = *(const float4*)(colp + 0);
  float4 b1 = *(const float4*)(colp + 4);
  float4 b2 = *(const float4*)(colp + 8);
  float4 b3 = *(const float4*)(colp + 12);
  frag2 r;
  split1(b0.x, r.hi, r.lo, 0);  split1(b0.y, r.hi, r.lo, 1);
  split1(b0.z, r.hi, r.lo, 2);  split1(b0.w, r.hi, r.lo, 3);
  split1(b1.x, r.hi, r.lo, 4);  split1(b1.y, r.hi, r.lo, 5);
  split1(b1.z, r.hi, r.lo, 6);  split1(b1.w, r.hi, r.lo, 7);
  split1(b2.x, r.hi, r.lo, 8);  split1(b2.y, r.hi, r.lo, 9);
  split1(b2.z, r.hi, r.lo, 10); split1(b2.w, r.hi, r.lo, 11);
  split1(b3.x, r.hi, r.lo, 12); split1(b3.y, r.hi, r.lo, 13);
  split1(b3.z, r.hi, r.lo, 14); split1(b3.w, r.hi, r.lo, 15);
  return r;
}

// packed f16 fragment (32-byte contiguous load)
__device__ __forceinline__ v16h load_frag_h(const _Float16* colp) {
  union { uint4 u[2]; v16h h; } t;
  t.u[0] = *(const uint4*)(colp);
  t.u[1] = *(const uint4*)(colp + 8);
  return t.h;
}

__device__ __forceinline__ v8f wmma_f16(v16h a, v16h b, v8f c) {
  return __builtin_amdgcn_wmma_f32_16x16x32_f16(false, a, false, b, (short)0, c, false, false);
}

// split-precision accumulate: acc += Ah*Bh + Ah*Bl + Al*Bh  (~fp32 accuracy)
__device__ __forceinline__ v8f wmma_split(const frag2& a, const frag2& b, v8f acc) {
  acc = wmma_f16(a.lo, b.hi, acc);
  acc = wmma_f16(a.hi, b.lo, acc);
  acc = wmma_f16(a.hi, b.hi, acc);
  return acc;
}

__device__ __forceinline__ float wave_sum(float v) {
#pragma unroll
  for (int o = 16; o > 0; o >>= 1) v += __shfl_xor(v, o, 32);
  return v;
}

// ---------- kernels ----------

// zero agg, deg = 1.0 (self-loop weight)
__global__ void tg_init_kernel(float* __restrict__ agg, float* __restrict__ deg, int n) {
  size_t i = (size_t)blockIdx.x * blockDim.x + threadIdx.x;
  if (i < (size_t)n * DIM) agg[i] = 0.0f;
  if (i < (size_t)n) deg[i] = 1.0f;
}

// one wave per node: score = dot(x[i], p) / ||p||, stored as monotonic uint key
__global__ void tg_score_kernel(const float* __restrict__ x, const float* __restrict__ p,
                                unsigned* __restrict__ key, int n) {
  int wid = blockIdx.x * (blockDim.x >> 5) + (threadIdx.x >> 5);
  if (wid >= n) return;
  int lane = threadIdx.x & 31;
  float4 xv = ((const float4*)(x + (size_t)wid * DIM))[lane];
  float4 pv = ((const float4*)p)[lane];
  float d  = xv.x * pv.x + xv.y * pv.y + xv.z * pv.z + xv.w * pv.w;
  float pp = pv.x * pv.x + pv.y * pv.y + pv.z * pv.z + pv.w * pv.w;
  d = wave_sum(d);
  pp = wave_sum(pp);
  if (lane == 0) key[wid] = key_of(d * rsqrtf(pp));
}

// single-block exact radix-select top-128 (desc, index-asc tiebreak) + build x_tilde
__global__ void tg_topk_kernel(const float* __restrict__ x, const unsigned* __restrict__ key,
                               float* __restrict__ xt, int n) {
  __shared__ unsigned sh_hist[256];
  __shared__ unsigned sh_selkey[256];
  __shared__ int      sh_selidx[256];
  __shared__ int      sh_perm[128];
  __shared__ float    sh_tv[128];
  __shared__ unsigned sh_prefix;
  __shared__ int      sh_kneed;
  __shared__ int      sh_cnt;

  int tid = threadIdx.x;
  int nt  = blockDim.x;
  if (tid == 0) { sh_prefix = 0u; sh_kneed = 128; }
  __syncthreads();

  for (int shift = 24; shift >= 0; shift -= 8) {
    for (int b = tid; b < 256; b += nt) sh_hist[b] = 0u;
    __syncthreads();
    unsigned pfx = sh_prefix;
    for (int i = tid; i < n; i += nt) {
      unsigned k = key[i];
      bool match = (shift == 24) ? true : (((k ^ pfx) >> (shift + 8)) == 0u);
      if (match) atomicAdd(&sh_hist[(k >> shift) & 255u], 1u);
    }
    __syncthreads();
    if (tid == 0) {
      int need = sh_kneed;
      int b = 255;
      for (; b > 0; --b) {
        int c = (int)sh_hist[b];
        if (need <= c) break;
        need -= c;
      }
      sh_kneed  = need;
      sh_prefix = pfx | (((unsigned)b) << shift);
    }
    __syncthreads();
  }
  unsigned T = sh_prefix;   // exact value of 128th-largest key

  if (tid == 0) sh_cnt = 0;
  __syncthreads();
  for (int i = tid; i < n; i += nt) {
    unsigned k = key[i];
    if (k >= T) {
      int p = atomicAdd(&sh_cnt, 1);
      if (p < 256) { sh_selkey[p] = k; sh_selidx[p] = i; }
    }
  }
  __syncthreads();
  int cnt = sh_cnt; if (cnt > 256) cnt = 256;
  for (int i = tid; i < 256; i += nt)
    if (i >= cnt) { sh_selkey[i] = 0u; sh_selidx[i] = 0x7FFFFFFF; }
  __syncthreads();

  // bitonic sort 256 entries: key desc, index asc on ties
  for (unsigned k2 = 2; k2 <= 256; k2 <<= 1) {
    for (unsigned j = k2 >> 1; j > 0; j >>= 1) {
      if (tid < 256) {
        unsigned i = (unsigned)tid, ixj = i ^ j;
        if (ixj > i) {
          unsigned ka = sh_selkey[i], kb = sh_selkey[ixj];
          int ia = sh_selidx[i], ib = sh_selidx[ixj];
          bool aBefore = (ka > kb) || (ka == kb && ia < ib);
          bool dirDesc = ((i & k2) == 0);
          bool doSwap = dirDesc ? !aBefore : aBefore;
          if (doSwap) {
            sh_selkey[i] = kb; sh_selkey[ixj] = ka;
            sh_selidx[i] = ib; sh_selidx[ixj] = ia;
          }
        }
      }
      __syncthreads();
    }
  }

  if (tid < 128) {
    sh_perm[tid] = sh_selidx[tid];
    sh_tv[tid]   = tanhf(key_to_float(sh_selkey[tid]));
  }
  __syncthreads();

  // x_tilde[r][c] = x[perm[r]][c] * tanh(val[r])
  for (int i = tid; i < DIM * DIM; i += nt) {
    int r = i >> 7, c = i & 127;
    xt[i] = x[(size_t)sh_perm[r] * DIM + c] * sh_tv[r];
  }
}

// GRU gate GEMMs: gi = xt @ w_ih^T, gh = W0 @ w_hh^T  (each 128x384, K=128)
// split-precision WMMA for ~fp32 accuracy
__global__ void tg_gru_gemm_kernel(const float* __restrict__ xt, const float* __restrict__ W0,
                                   const float* __restrict__ w_ih, const float* __restrict__ w_hh,
                                   float* __restrict__ gi, float* __restrict__ gh) {
  int wid = blockIdx.x * (blockDim.x >> 5) + (threadIdx.x >> 5);
  if (wid >= 2 * 8 * 24) return;
  int mat = wid / 192;
  int rem = wid % 192;
  int rt = rem / 24, ct = rem % 24;
  const float* A = mat ? W0 : xt;
  const float* B = mat ? w_hh : w_ih;      // [384,128] row-major: row g holds K
  float* O = mat ? gh : gi;

  int lane = threadIdx.x & 31;
  int m = lane & 15, half = lane >> 4;

  v8f acc;
#pragma unroll
  for (int i = 0; i < 8; ++i) acc[i] = 0.0f;
  const float* arow = A + (size_t)(rt * 16 + m) * DIM;
  const float* bcol = B + (size_t)(ct * 16 + m) * DIM;   // B-lane column = lane&15
#pragma unroll
  for (int ks = 0; ks < 4; ++ks) {
    int kbase = ks * 32;
    frag2 af = load_a_frag_split(arow + kbase, half);
    frag2 bf = load_b_frag_split(bcol + kbase + half * 16);
    acc = wmma_split(af, bf, acc);
  }
  // D layout: VGPR v -> row rt*16 + v + 8*half, col ct*16 + (lane&15)
  float* orow = O + (size_t)(rt * 16 + half * 8) * GDIM + ct * 16 + m;
#pragma unroll
  for (int v = 0; v < 8; ++v) orow[(size_t)v * GDIM] = acc[v];
}

// GRU elementwise: W = (1-z)*n + z*W0 ; emit W transposed as two f16 planes (hi/lo)
__global__ void tg_gru_elem_kernel(const float* __restrict__ gi, const float* __restrict__ gh,
                                   const float* __restrict__ b_ih, const float* __restrict__ b_hh,
                                   const float* __restrict__ W0,
                                   _Float16* __restrict__ Wt_hi, _Float16* __restrict__ Wt_lo) {
  int i = blockIdx.x * blockDim.x + threadIdx.x;
  if (i >= DIM * DIM) return;
  int r = i >> 7, c = i & 127;
  const float* gir = gi + (size_t)r * GDIM;
  const float* ghr = gh + (size_t)r * GDIM;
  float ir = gir[c]       + b_ih[c],       hr = ghr[c]       + b_hh[c];
  float iz = gir[128 + c] + b_ih[128 + c], hz = ghr[128 + c] + b_hh[128 + c];
  float in = gir[256 + c] + b_ih[256 + c], hn = ghr[256 + c] + b_hh[256 + c];
  float rg = 1.0f / (1.0f + expf(-(ir + hr)));
  float zg = 1.0f / (1.0f + expf(-(iz + hz)));
  float ng = tanhf(in + rg * hn);
  float wv = (1.0f - zg) * ng + zg * W0[i];
  _Float16 hh = (_Float16)wv;
  Wt_hi[(size_t)c * DIM + r] = hh;                          // Wt[n][k] = W[k][n]
  Wt_lo[(size_t)c * DIM + r] = (_Float16)(wv - (float)hh);
}

// xw = x @ W : wave per 16-row strip, all 128 columns (8 accumulator tiles),
// split-precision WMMA (Ah*Bh + Ah*Bl + Al*Bh)
__global__ void tg_xw_gemm_kernel(const float* __restrict__ x,
                                  const _Float16* __restrict__ Wt_hi,
                                  const _Float16* __restrict__ Wt_lo,
                                  float* __restrict__ xw, int nrows) {
  int wid = blockIdx.x * (blockDim.x >> 5) + (threadIdx.x >> 5);
  int r0 = wid * 16;
  if (r0 >= nrows) return;
  int lane = threadIdx.x & 31;
  int m = lane & 15, half = lane >> 4;

  v8f acc[8];
#pragma unroll
  for (int t = 0; t < 8; ++t)
#pragma unroll
    for (int i = 0; i < 8; ++i) acc[t][i] = 0.0f;

  const float* arow = x + (size_t)(r0 + m) * DIM;
#pragma unroll
  for (int ks = 0; ks < 4; ++ks) {
    int kbase = ks * 32;
    frag2 af = load_a_frag_split(arow + kbase, half);
#pragma unroll
    for (int ct = 0; ct < 8; ++ct) {
      size_t boff = (size_t)(ct * 16 + m) * DIM + kbase + half * 16;
      frag2 bf;
      bf.hi = load_frag_h(Wt_hi + boff);
      bf.lo = load_frag_h(Wt_lo + boff);
      acc[ct] = wmma_split(af, bf, acc[ct]);
    }
  }
#pragma unroll
  for (int ct = 0; ct < 8; ++ct) {
    float* orow = xw + (size_t)(r0 + half * 8) * DIM + ct * 16 + m;
#pragma unroll
    for (int v = 0; v < 8; ++v) orow[(size_t)v * DIM] = acc[ct][v];
  }
}

// deg[col] += edge_weight (self-loop 1.0 pre-initialized)
__global__ void tg_deg_kernel(const int* __restrict__ ei, const float* __restrict__ ew,
                              float* __restrict__ deg, int E) {
  int e = blockIdx.x * blockDim.x + threadIdx.x;
  if (e >= E) return;
  unsafeAtomicAdd(&deg[ei[E + e]], ew[e]);
}

__global__ void tg_dinv_kernel(const float* __restrict__ deg, float* __restrict__ dinv, int n) {
  int i = blockIdx.x * blockDim.x + threadIdx.x;
  if (i >= n) return;
  float d = deg[i];
  dinv[i] = (d > 0.0f) ? rsqrtf(d) : 0.0f;
}

// one wave per edge: agg[col] += norm * xw[row]
__global__ void tg_agg_kernel(const int* __restrict__ ei, const float* __restrict__ ew,
                              const float* __restrict__ dinv, const float* __restrict__ xw,
                              float* __restrict__ agg, int E) {
  int wid = blockIdx.x * (blockDim.x >> 5) + (threadIdx.x >> 5);
  if (wid >= E) return;
  int lane = threadIdx.x & 31;
  int row = ei[wid];
  int col = ei[E + wid];
  float nrm = dinv[row] * ew[wid] * dinv[col];
  float4 v = ((const float4*)(xw + (size_t)row * DIM))[lane];
  float* ag = agg + (size_t)col * DIM + lane * 4;
  unsafeAtomicAdd(ag + 0, v.x * nrm);
  unsafeAtomicAdd(ag + 1, v.y * nrm);
  unsafeAtomicAdd(ag + 2, v.z * nrm);
  unsafeAtomicAdd(ag + 3, v.w * nrm);
}

// self-loop + relu + linear head: one wave per node
__global__ void tg_head_kernel(const float* __restrict__ agg, const float* __restrict__ xw,
                               const float* __restrict__ dinv, const float* __restrict__ w_lin,
                               const float* __restrict__ b_lin, float* __restrict__ out, int n) {
  int wid = blockIdx.x * (blockDim.x >> 5) + (threadIdx.x >> 5);
  if (wid >= n) return;
  int lane = threadIdx.x & 31;
  float4 a  = ((const float4*)(agg + (size_t)wid * DIM))[lane];
  float4 xv = ((const float4*)(xw  + (size_t)wid * DIM))[lane];
  float di = dinv[wid];
  float s = di * di;
  a.x = fmaxf(a.x + s * xv.x, 0.0f);
  a.y = fmaxf(a.y + s * xv.y, 0.0f);
  a.z = fmaxf(a.z + s * xv.z, 0.0f);
  a.w = fmaxf(a.w + s * xv.w, 0.0f);
  float4 wl = ((const float4*)w_lin)[lane];
  float d = a.x * wl.x + a.y * wl.y + a.z * wl.z + a.w * wl.w;
  d = wave_sum(d);
  if (lane == 0) out[wid] = d + b_lin[0];
}

// ---------- launch ----------

extern "C" void kernel_launch(void* const* d_in, const int* in_sizes, int n_in,
                              void* d_out, int out_size, void* d_ws, size_t ws_size,
                              hipStream_t stream) {
  const float* x     = (const float*)d_in[0];
  const int*   ei    = (const int*)d_in[1];
  const float* ew    = (const float*)d_in[2];
  const float* p     = (const float*)d_in[3];
  const float* W0    = (const float*)d_in[4];
  const float* w_ih  = (const float*)d_in[5];
  const float* w_hh  = (const float*)d_in[6];
  const float* b_ih  = (const float*)d_in[7];
  const float* b_hh  = (const float*)d_in[8];
  const float* w_lin = (const float*)d_in[9];
  const float* b_lin = (const float*)d_in[10];
  float* out = (float*)d_out;

  const int N = in_sizes[0] / DIM;
  const int E = in_sizes[2];

  char* base = (char*)d_ws;
  size_t off = 0;
  auto carve = [&](size_t bytes) -> void* {
    void* pp = base + off;
    off = (off + bytes + 255) & ~(size_t)255;
    return pp;
  };
  unsigned*  key   = (unsigned*)carve((size_t)N * 4);
  float*     deg   = (float*)carve((size_t)N * 4);
  float*     dinv  = (float*)carve((size_t)N * 4);
  float*     xt    = (float*)carve((size_t)DIM * DIM * 4);
  float*     gi    = (float*)carve((size_t)DIM * GDIM * 4);
  float*     gh    = (float*)carve((size_t)DIM * GDIM * 4);
  _Float16*  Wt_hi = (_Float16*)carve((size_t)DIM * DIM * 2);
  _Float16*  Wt_lo = (_Float16*)carve((size_t)DIM * DIM * 2);
  float*     xw    = (float*)carve((size_t)N * DIM * 4);
  float*     agg   = (float*)carve((size_t)N * DIM * 4);

  const int TB = 256;            // 8 waves per block
  const int WPB = TB / 32;

  // 1. init agg/deg
  {
    size_t tot = (size_t)N * DIM;
    int blocks = (int)((tot + TB - 1) / TB);
    tg_init_kernel<<<blocks, TB, 0, stream>>>(agg, deg, N);
  }
  // 2. scores -> keys
  tg_score_kernel<<<(N + WPB - 1) / WPB, TB, 0, stream>>>(x, p, key, N);
  // 3. top-k + x_tilde
  tg_topk_kernel<<<1, 1024, 0, stream>>>(x, key, xt, N);
  // 4. GRU gate GEMMs (split-precision WMMA): 384 waves
  tg_gru_gemm_kernel<<<(384 + WPB - 1) / WPB, TB, 0, stream>>>(xt, W0, w_ih, w_hh, gi, gh);
  // 5. GRU elementwise -> Wt hi/lo (f16, transposed)
  tg_gru_elem_kernel<<<(DIM * DIM + TB - 1) / TB, TB, 0, stream>>>(gi, gh, b_ih, b_hh, W0,
                                                                   Wt_hi, Wt_lo);
  // 6. xw = x @ W (split-precision WMMA): one wave per 16 rows
  {
    int nwaves = (N + 15) / 16;
    tg_xw_gemm_kernel<<<(nwaves + WPB - 1) / WPB, TB, 0, stream>>>(x, Wt_hi, Wt_lo, xw, N);
  }
  // 7. degree accumulation + normalization
  tg_deg_kernel<<<(E + TB - 1) / TB, TB, 0, stream>>>(ei, ew, deg, E);
  tg_dinv_kernel<<<(N + TB - 1) / TB, TB, 0, stream>>>(deg, dinv, N);
  // 8. edge scatter-add
  tg_agg_kernel<<<(E + WPB - 1) / WPB, TB, 0, stream>>>(ei, ew, dinv, xw, agg, E);
  // 9. self-loop + relu + linear head
  tg_head_kernel<<<(N + WPB - 1) / WPB, TB, 0, stream>>>(agg, xw, dinv, w_lin, b_lin, out, N);
}
